// Synthesis_7945689498064
// MI455X (gfx1250) — compile-verified, matrix operand order checked by reading, not run on hardware
//
#include <hip/hip_runtime.h>
#include <cmath>

typedef __attribute__((ext_vector_type(16))) _Float16 v16h;
typedef __attribute__((ext_vector_type(8)))  _Float16 v8h;
typedef __attribute__((ext_vector_type(8)))  float    v8f;

#define NLAYER   15     // NUM_LAYERS + 1
#define WS_SLOTS 16     // NUM_LAYERS + 2
#define SDIM     512
#define NB       2
#define IMG      256
#define CLV      256.0f

// ----------------------------- device kernels -----------------------------

__global__ void k_styles(const float* __restrict__ wsrc, const float* __restrict__ aw,
                         const float* __restrict__ ab, float* __restrict__ s,
                         int Cin, int wIdx) {
  int idx = blockIdx.x * blockDim.x + threadIdx.x;
  if (idx >= NB * Cin) return;
  int b = idx / Cin, c = idx - b * Cin;
  const float* wv = wsrc + ((size_t)b * WS_SLOTS + wIdx) * SDIM;
  const float* ar = aw + (size_t)c * SDIM;
  float acc = 0.f;
  for (int t = 0; t < SDIM; ++t) acc += wv[t] * ar[t];
  s[idx] = acc * 0.04419417382415922f + ab[c];  // 1/sqrt(512)
}

// modulate + (optional) demod, write f16 weights [NB][CoPad][Kpad], K = ci*kk + kidx
__global__ void k_modw(const float* __restrict__ w, const float* __restrict__ s,
                       _Float16* __restrict__ wt, int Cout, int Cin, int kk,
                       int CoPad, int Kpad, int demod) {
  int co = blockIdx.x, b = blockIdx.y, tid = threadIdx.x;
  int Ktot = Cin * kk;
  _Float16* dst = wt + ((size_t)b * CoPad + co) * Kpad;
  if (co >= Cout) {                       // zero pad rows (uniform per block)
    for (int i = tid; i < Kpad; i += 256) dst[i] = (_Float16)0.f;
    return;
  }
  float scale = rsqrtf((float)Ktot);
  const float* wr = w + (size_t)co * Ktot;
  const float* sb = s ? s + (size_t)b * Cin : nullptr;
  float ss = 0.f;
  for (int i = tid; i < Ktot; i += 256) {
    float sv = sb ? sb[i / kk] : 1.f;
    float v = wr[i] * sv * scale;
    ss += v * v;
  }
  __shared__ float red[256];
  red[tid] = ss; __syncthreads();
  for (int st = 128; st > 0; st >>= 1) {
    if (tid < st) red[tid] += red[tid + st];
    __syncthreads();
  }
  float d = demod ? rsqrtf(red[0] + 1e-8f) : 1.f;
  for (int i = tid; i < Kpad; i += 256) {
    float v = 0.f;
    if (i < Ktot) { float sv = sb ? sb[i / kk] : 1.f; v = wr[i] * sv * scale * d; }
    dst[i] = (_Float16)v;
  }
}

// Cooperative implicit-GEMM conv, register-blocked 4x4 (M x N fragments).
// Block = 256 threads (8 waves) owns a 64-pixel N-tile and ALL output channels.
// Per 32-wide K chunk: stage 4 B fragments once into LDS (pre-permuted to the
// WMMA B-fragment layout), barrier, then each wave fires up to 4x4 WMMAs,
// reusing each A fragment across the 4 N fragments. LDS double-buffered ->
// one barrier per chunk.
template <int KSZ>
__global__ __launch_bounds__(256) void k_conv_wmma(
    const float* __restrict__ x, const _Float16* __restrict__ wt,
    const float* __restrict__ bias, float* __restrict__ y,
    int Cin, int H, int W, int Cout, int CoPad, int pad,
    int Ho, int Wo, int Kpad) {
  const int tid = threadIdx.x;
  const int wave = tid >> 5;
  const int lane = tid & 31;
  const int laneHi = lane >> 4;
  const int lm = lane & 15;
  const int hi8 = laneHi ? 8 : 0;
  const int b = blockIdx.z;
  const int n0 = blockIdx.x * 64;
  const int P = Ho * Wo;
  const int kk = KSZ * KSZ;
  const int Ktot = Cin * kk;
  const float* xb = x + (size_t)b * Cin * H * W;
  const _Float16* wb = wt + (size_t)b * CoPad * Kpad;

  __shared__ __align__(32) _Float16 lb[2][4][512];   // [buf][nfrag][lane*16 + j]

  v8f acc[4][4] = {};                                // [mtile][nfrag]
  int buf = 0;
  for (int kb = 0; kb < Kpad; kb += 32, buf ^= 1) {
    // ---- stage 4 B fragments: 2048 halves, 8 per thread
    for (int s = tid; s < 2048; s += 256) {
      int nf = s >> 9;
      int sl = (s >> 4) & 31, j = s & 15;
      int slHi8 = (sl & 16) ? 8 : 0;
      int Kj = kb + ((j < 8) ? (j + slHi8) : (16 + (j - 8) + slHi8));
      int p = n0 + nf * 16 + (sl & 15);
      float v = 0.f;
      if (p < P && Kj < Ktot) {
        int py = p / Wo;
        int px = p - py * Wo;
        int ci = Kj / kk;                  // KSZ constexpr -> magic-number division
        int kr = Kj - ci * kk;
        int ky = kr / KSZ;
        int kx = kr - ky * KSZ;
        int iy = py + ky - pad;
        int ix = px + kx - pad;
        if ((unsigned)iy < (unsigned)H && (unsigned)ix < (unsigned)W)
          v = xb[((size_t)ci * H + iy) * W + ix];
      }
      lb[buf][nf][(sl << 4) + j] = (_Float16)v;
    }
    __syncthreads();
    v16h bm[4];
#pragma unroll
    for (int nf = 0; nf < 4; ++nf)
      bm[nf] = *(const v16h*)&lb[buf][nf][lane * 16];
#pragma unroll
    for (int t = 0; t < 4; ++t) {
      int m0t = (wave + 8 * t) * 16;
      if (m0t < CoPad) {                   // wave-uniform guard, EXEC stays full
        const _Float16* wr = wb + (size_t)(m0t + lm) * Kpad + kb + hi8;
        v8h alo = *(const v8h*)wr;
        v8h ahi = *(const v8h*)(wr + 16);
        v16h a;
#pragma unroll
        for (int j = 0; j < 8; ++j) { a[j] = alo[j]; a[j + 8] = ahi[j]; }
#pragma unroll
        for (int nf = 0; nf < 4; ++nf)     // A fragment reused 4x
          acc[t][nf] = __builtin_amdgcn_wmma_f32_16x16x32_f16(
              false, a, false, bm[nf], (short)0, acc[t][nf], false, false);
      }
    }
  }
  // ---- epilogue: f32 16x16 C layout -> [b][co][p], fused bias
#pragma unroll
  for (int t = 0; t < 4; ++t) {
    int m0t = (wave + 8 * t) * 16;
    if (m0t < CoPad) {
#pragma unroll
      for (int nf = 0; nf < 4; ++nf) {
        int p = n0 + nf * 16 + lm;
        if (p < P) {
#pragma unroll
          for (int v = 0; v < 8; ++v) {
            int co = m0t + v + hi8;
            if (co < Cout) {
              float val = acc[t][nf][v];
              if (bias) val += bias[co];
              y[((size_t)b * Cout + co) * P + p] = val;
            }
          }
        }
      }
    }
  }
}

// synthesis-input prep: per (b,c) -> fr0, fr1, phase, amp
__global__ void k_input_prep(const float* __restrict__ wsrc, const float* __restrict__ aw,
                             const float* __restrict__ ab, const float* __restrict__ freqs,
                             const float* __restrict__ phases, float* __restrict__ prep,
                             int C, float sr, float bw) {
  int idx = blockIdx.x * blockDim.x + threadIdx.x;
  if (idx >= NB * C) return;
  int b = idx / C, c = idx - b * C;
  const float* wv = wsrc + (size_t)b * WS_SLOTS * SDIM;  // ws[:,0]
  float t[4];
  for (int j = 0; j < 4; ++j) {
    const float* ar = aw + (size_t)j * SDIM;
    float acc = 0.f;
    for (int k = 0; k < SDIM; ++k) acc += wv[k] * ar[k];
    t[j] = acc * 0.04419417382415922f + ab[j];
  }
  float nrm = sqrtf(t[0] * t[0] + t[1] * t[1]);
  for (int j = 0; j < 4; ++j) t[j] /= nrm;
  float cr = t[0], sn = t[1];
  float tv0 = cr * (-t[2]) + sn * t[3];    // rot row0 . [-t2,-t3]
  float tv1 = sn * (-t[2]) + cr * (-t[3]); // rot row1 . [-t2,-t3]
  float f0 = freqs[(size_t)c * 2], f1 = freqs[(size_t)c * 2 + 1];
  float ph = phases[c] + f0 * tv0 + f1 * tv1;
  float fr0 = f0 * cr + f1 * sn;           // freqs @ rot
  float fr1 = -f0 * sn + f1 * cr;
  float amp = 1.f - (sqrtf(fr0 * fr0 + fr1 * fr1) - bw) / (sr * 0.5f - bw);
  amp = fminf(fmaxf(amp, 0.f), 1.f);
  float* pr = prep + (size_t)idx * 4;
  pr[0] = fr0; pr[1] = fr1; pr[2] = ph; pr[3] = amp;
}

__global__ void k_feat(const float* __restrict__ prep, float* __restrict__ out,
                       int C, int N, float sr) {
  long long idx = (long long)blockIdx.x * 256 + threadIdx.x;
  long long total = (long long)NB * C * N * N;
  if (idx >= total) return;
  int x = (int)(idx % N); long long r = idx / N;
  int y = (int)(r % N); r /= N;
  int c = (int)(r % C); int b = (int)(r / C);
  const float* pr = prep + ((size_t)b * C + c) * 4;
  float bx = ((x + 0.5f) * 2.f / N - 1.f) * (0.5f * N / sr);
  float by = ((y + 0.5f) * 2.f / N - 1.f) * (0.5f * N / sr);
  float arg = bx * pr[0] + by * pr[1] + pr[2];
  out[idx] = sinf(arg * 6.28318530717958647692f) * pr[3];
}

// horizontal upfirdn pass (zero-stuff by `up`, pad p0, T taps) -> [NB][C][Hp][W1]
__global__ void k_up_h(const float* __restrict__ in, float* __restrict__ out,
                       int C, int H, int W, int up, int p0, int T,
                       const float* __restrict__ f, float fscale, int Hp, int W1) {
  long long idx = (long long)blockIdx.x * 256 + threadIdx.x;
  long long total = (long long)NB * C * Hp * W1;
  if (idx >= total) return;
  int x1 = (int)(idx % W1); long long r = idx / W1;
  int y = (int)(r % Hp); r /= Hp;
  int c = (int)(r % C); int b = (int)(r / C);
  float acc = 0.f;
  int yy = y - p0;
  if (yy >= 0 && yy % up == 0) {
    int iy = yy / up;
    if (iy < H) {
      const float* row = in + (((size_t)b * C + c) * H + iy) * W;
      for (int t = 0; t < T; ++t) {
        int xu = x1 + t - p0;
        if (xu >= 0 && xu % up == 0) {
          int ix = xu / up;
          if (ix < W) acc += f[t] * row[ix];
        }
      }
      acc *= fscale;
    }
  }
  out[idx] = acc;
}

// vertical upfirdn pass + leaky-relu*gain + clamp -> [NB][C][H2][W1]
__global__ void k_up_v_lrelu(const float* __restrict__ in, float* __restrict__ out,
                             int C, int Hp, int W1, int T, const float* __restrict__ f,
                             float fscale, float gain, float slope) {
  int H2 = Hp - T + 1;
  long long idx = (long long)blockIdx.x * 256 + threadIdx.x;
  long long total = (long long)NB * C * H2 * W1;
  if (idx >= total) return;
  int x = (int)(idx % W1); long long r = idx / W1;
  int y2 = (int)(r % H2); r /= H2;
  int c = (int)(r % C); int b = (int)(r / C);
  const float* col = in + (((size_t)b * C + c) * Hp + y2) * W1 + x;
  float acc = 0.f;
  for (int t = 0; t < T; ++t) acc += f[t] * col[(size_t)t * W1];
  float v = acc * fscale;
  v = v > 0.f ? v : v * slope;
  v *= gain;
  v = fminf(fmaxf(v, -CLV), CLV);
  out[idx] = v;
}

// radial (2D) down-filter + stride-`dn` decimation: [NB][C][Hi][Hi] -> [NB][C][Ho][Ho]
__global__ void k_down_rad(const float* __restrict__ in, float* __restrict__ out,
                           int C, int Hi, int T, const float* __restrict__ f2,
                           int dn, int Ho) {
  long long idx = (long long)blockIdx.x * 256 + threadIdx.x;
  long long total = (long long)NB * C * Ho * Ho;
  if (idx >= total) return;
  int xo = (int)(idx % Ho); long long r = idx / Ho;
  int yo = (int)(r % Ho); r /= Ho;
  int c = (int)(r % C); int b = (int)(r / C);
  const float* base = in + ((size_t)b * C + c) * Hi * Hi;
  int y0 = yo * dn, x0 = xo * dn;
  float acc = 0.f;
  for (int dy = 0; dy < T; ++dy) {
    int iy = y0 + dy;
    if (iy >= Hi) break;
    const float* row = base + (size_t)iy * Hi;
    const float* fr = f2 + (size_t)dy * T;
    for (int dx = 0; dx < T; ++dx) {
      int ix = x0 + dx;
      if (ix < Hi) acc += fr[dx] * row[ix];
    }
  }
  out[idx] = acc;
}

// separable down: horizontal taps with stride dn -> [NB][C][Hi][Wo]
__global__ void k_down_sep_h(const float* __restrict__ in, float* __restrict__ out,
                             int C, int Hi, int T, const float* __restrict__ f,
                             int dn, int Wo) {
  long long idx = (long long)blockIdx.x * 256 + threadIdx.x;
  long long total = (long long)NB * C * Hi * Wo;
  if (idx >= total) return;
  int xo = (int)(idx % Wo); long long r = idx / Wo;
  int y = (int)(r % Hi); r /= Hi;
  int c = (int)(r % C); int b = (int)(r / C);
  const float* row = in + (((size_t)b * C + c) * Hi + y) * Hi;
  int x0 = xo * dn;
  float acc = 0.f;
  for (int t = 0; t < T; ++t) {
    int ix = x0 + t;
    if (ix < Hi) acc += f[t] * row[ix];
  }
  out[idx] = acc;
}

// separable down: vertical taps with stride dn -> [NB][C][Ho][Wo]
__global__ void k_down_sep_v(const float* __restrict__ in, float* __restrict__ out,
                             int C, int Hi, int Wo, int T, const float* __restrict__ f,
                             int dn, int Ho) {
  long long idx = (long long)blockIdx.x * 256 + threadIdx.x;
  long long total = (long long)NB * C * Ho * Wo;
  if (idx >= total) return;
  int xo = (int)(idx % Wo); long long r = idx / Wo;
  int yo = (int)(r % Ho); r /= Ho;
  int c = (int)(r % C); int b = (int)(r / C);
  const float* base = in + ((size_t)b * C + c) * (size_t)Hi * Wo + xo;
  int y0 = yo * dn;
  float acc = 0.f;
  for (int t = 0; t < T; ++t) {
    int iy = y0 + t;
    if (iy < Hi) acc += f[t] * base[(size_t)iy * Wo];
  }
  out[idx] = acc;
}

// rgb layer epilogue: clamp then * 0.25 into d_out
__global__ void k_rgb_final(const float* __restrict__ in, float* __restrict__ out) {
  long long idx = (long long)blockIdx.x * 256 + threadIdx.x;
  long long total = (long long)NB * 3 * IMG * IMG;
  if (idx >= total) return;
  float v = in[idx];
  v = fminf(fmaxf(v, -CLV), CLV);
  out[idx] = v * 0.25f;
}

// ----------------------------- host config --------------------------------

namespace {

struct LCfg {
  int in_ch, out_ch, k, is_rgb, is_crit, up, down, p0, p1, upT, dnT, in_size, out_size;
  float gain, slope;
  int fuOff, fdOff;
};

inline int floordiv(int a, int b) {
  int q = a / b;
  if ((a % b != 0) && ((a < 0) != (b < 0))) --q;
  return q;
}

double bessel_i0(double x) {
  double s = 1.0, t = 1.0, x2 = x * x * 0.25;
  for (int k = 1; k < 64; ++k) {
    t *= x2 / ((double)k * k);
    s += t;
    if (t < 1e-18 * s) break;
  }
  return s;
}

double bessel_j1(double x) {
  double ax = fabs(x), y = x * x;
  double p1 = x * (72362614232.0 + y * (-7895059235.0 + y * (242396853.1 +
              y * (-2972611.439 + y * (15704.4826 + y * -30.16036606)))));
  double q1 = 144725228442.0 + y * (2300535178.0 + y * (18583304.74 +
              y * (99447.43394 + y * (376.9991397 + y))));
  double small = p1 / q1;
  double axs = fmax(ax, 1e-30);
  double z = 8.0 / axs, y2 = z * z;
  double xx = ax - 2.356194491;
  double p2 = 1.0 + y2 * (0.00183105 + y2 * (-3.516396496e-05 +
              y2 * (2.457520174e-06 + y2 * -2.40337019e-07)));
  double q2 = 0.04687499995 + y2 * (-0.0002002690873 + y2 * (8.449199096e-06 +
              y2 * (-8.8228987e-07 + y2 * 1.05787412e-07)));
  double large = sqrt(0.636619772 / axs) * (cos(xx) * p2 - z * sin(xx) * q2) *
                 (x < 0 ? -1.0 : 1.0);
  return ax < 8.0 ? small : large;
}

double kaiser_beta(double a) {
  if (a > 50.0) return 0.1102 * (a - 8.7);
  if (a > 21.0) return 0.5842 * pow(a - 21.0, 0.4) + 0.07886 * (a - 21.0);
  return 0.0;
}

void design_filter(int nt, double cutoff, double width, double fs, bool radial, float* out) {
  if (nt == 1) { out[0] = 1.f; return; }
  double atten = 2.285 * (nt - 1) * M_PI * (width / (fs / 2.0)) + 7.95;
  double beta = kaiser_beta(atten);
  double win[32];
  double i0b = bessel_i0(beta);
  for (int n = 0; n < nt; ++n) {
    double rr = 2.0 * n / (nt - 1) - 1.0;
    win[n] = bessel_i0(beta * sqrt(fmax(0.0, 1.0 - rr * rr))) / i0b;
  }
  if (!radial) {
    double c = cutoff / (fs / 2.0), s = 0.0, tmp[32];
    for (int n = 0; n < nt; ++n) {
      double m = n - (nt - 1) / 2.0, xx = c * m;
      double sinc = (xx == 0.0) ? 1.0 : sin(M_PI * xx) / (M_PI * xx);
      tmp[n] = c * sinc * win[n];
      s += tmp[n];
    }
    for (int n = 0; n < nt; ++n) out[n] = (float)(tmp[n] / s);
  } else {
    static double tmp2[1024];
    double s = 0.0;
    for (int i = 0; i < nt; ++i) {
      double xi = (i - (nt - 1) / 2.0) / fs;
      for (int j = 0; j < nt; ++j) {
        double xj = (j - (nt - 1) / 2.0) / fs;
        double r = hypot(xi, xj);
        double v = bessel_j1(2.0 * cutoff * M_PI * r) / (M_PI * fmax(r, 1e-30));
        v *= win[i] * win[j];
        tmp2[i * nt + j] = v;
        s += v;
      }
    }
    for (int i = 0; i < nt * nt; ++i) out[i] = (float)(tmp2[i] / s);
  }
}

inline unsigned gblk(long long n) { return (unsigned)((n + 255) / 256); }

} // namespace

// ----------------------------- launcher ------------------------------------

extern "C" void kernel_launch(void* const* d_in, const int* in_sizes, int n_in,
                              void* d_out, int out_size, void* d_ws, size_t ws_size,
                              hipStream_t stream) {
  (void)in_sizes; (void)n_in; (void)out_size; (void)ws_size;

  // ---- inputs (dict order of setup_inputs) ----
  const float* ws_in = (const float*)d_in[0];
  const float* iaw   = (const float*)d_in[1];
  const float* iab   = (const float*)d_in[2];
  const float* ifr   = (const float*)d_in[3];
  const float* iph   = (const float*)d_in[4];
  const float* iwt   = (const float*)d_in[5];
  const float* aff_w[NLAYER]; const float* aff_b[NLAYER];
  const float* conv_w[NLAYER]; const float* conv_b[NLAYER];
  for (int i = 0; i < NLAYER; ++i) {
    aff_w[i]  = (const float*)d_in[6 + i];
    aff_b[i]  = (const float*)d_in[6 + NLAYER + i];
    conv_w[i] = (const float*)d_in[6 + 2 * NLAYER + i];
    conv_b[i] = (const float*)d_in[6 + 3 * NLAYER + i];
  }

  // ---- replicate _make_config() (deterministic, double precision) ----
  double cut[NLAYER], stop[NLAYER], srate[NLAYER], hws[NLAYER], sz[NLAYER], ch[NLAYER];
  const double last_stop = 128.0 * pow(2.0, 0.3);
  const double first_stop = pow(2.0, 2.1);
  for (int i = 0; i < NLAYER; ++i) {
    double e = fmin((double)i / 12.0, 1.0);
    cut[i] = 2.0 * pow(64.0, e);
    stop[i] = first_stop * pow(last_stop / first_stop, e);
    srate[i] = exp2(ceil(log2(fmin(stop[i] * 2.0, 256.0))));
    hws[i] = fmax(stop[i], srate[i] / 2.0) - cut[i];
    sz[i] = srate[i] + 20.0;
    ch[i] = nearbyint(fmin(4096.0 / cut[i], 512.0));
  }
  sz[NLAYER - 2] = 256.0; sz[NLAYER - 1] = 256.0; ch[NLAYER - 1] = 3.0;

  static float h_filt[8192];
  LCfg L[NLAYER];
  int fpos = 0;
  for (int i = 0; i < NLAYER; ++i) {
    int p = i > 0 ? i - 1 : 0;
    LCfg& c = L[i];
    c.is_rgb = (i == NLAYER - 1);
    c.is_crit = (i >= NLAYER - 3);
    c.k = c.is_rgb ? 1 : 3;
    double fs = fmax(srate[p], srate[i]) * (c.is_rgb ? 1.0 : 2.0);
    c.up = (int)llround(fs / srate[p]);
    c.down = (int)llround(fs / srate[i]);
    c.upT = (c.up > 1 && !c.is_rgb) ? 6 * c.up : 1;
    c.dnT = (c.down > 1 && !c.is_rgb) ? 6 * c.down : 1;
    c.in_ch = (int)ch[p]; c.out_ch = (int)ch[i];
    c.in_size = (int)sz[p]; c.out_size = (int)sz[i];
    int pad_total = (c.out_size - 1) * c.down + 1 - (c.in_size + c.k - 1) * c.up +
                    c.upT + c.dnT - 2;
    c.p0 = floordiv(pad_total + c.up, 2);
    c.p1 = pad_total - c.p0;
    c.gain = c.is_rgb ? 1.f : (float)sqrt(2.0);
    c.slope = c.is_rgb ? 1.f : 0.2f;
    if (c.upT > 1) {
      design_filter(c.upT, cut[p], hws[p] * 2.0, fs, false, h_filt + fpos);
      c.fuOff = fpos; fpos += c.upT;
    } else c.fuOff = -1;
    if (c.dnT > 1) {
      bool rad = !c.is_crit;
      design_filter(c.dnT, cut[i], hws[i] * 2.0, fs, rad, h_filt + fpos);
      c.fdOff = fpos; fpos += rad ? c.dnT * c.dnT : c.dnT;
    } else c.fdOff = -1;
  }
  const int C0 = (int)ch[0];
  const int N0 = (int)sz[0];
  const float sr0 = (float)srate[0], bw0 = (float)cut[0];

  // ---- scratch carve ----
  long long maxBuf = (long long)NB * C0 * N0 * N0;  // feat / x0
  for (int i = 0; i < NLAYER; ++i) {
    const LCfg& c = L[i];
    int Hc = c.in_size + (c.k == 3 ? 2 : 0);
    long long convE = (long long)NB * c.out_ch * Hc * Hc;
    if (convE > maxBuf) maxBuf = convE;
    if (!c.is_rgb) {
      int Hp = Hc * c.up + c.p0 + c.p1;
      int W1 = Hp - c.upT + 1;
      long long e1 = (long long)NB * c.out_ch * Hp * W1;
      long long e2 = (long long)NB * c.out_ch * W1 * W1;
      long long eo = (long long)NB * c.out_ch * c.out_size * c.out_size;
      long long et = (long long)NB * c.out_ch * W1 * c.out_size;
      if (e1 > maxBuf) maxBuf = e1;
      if (e2 > maxBuf) maxBuf = e2;
      if (eo > maxBuf) maxBuf = eo;
      if (et > maxBuf) maxBuf = et;
    }
  }
  size_t off = 0;
  auto carve = [&](size_t bytes) { size_t o = off; off = (off + bytes + 255) & ~(size_t)255; return o; };
  size_t OFF_FILT = carve((size_t)fpos * sizeof(float));
  size_t OFF_PREP = carve((size_t)NB * C0 * 4 * sizeof(float));
  size_t OFF_STY  = carve((size_t)NB * 512 * sizeof(float));
  size_t OFF_WT   = carve((size_t)NB * 512 * 4608 * sizeof(_Float16));
  size_t OFF_R0   = carve((size_t)maxBuf * sizeof(float));
  size_t OFF_R1   = carve((size_t)maxBuf * sizeof(float));
  size_t OFF_R2   = carve((size_t)maxBuf * sizeof(float));

  char* base = (char*)d_ws;
  float*     d_filt = (float*)(base + OFF_FILT);
  float*     d_prep = (float*)(base + OFF_PREP);
  float*     d_sty  = (float*)(base + OFF_STY);
  _Float16*  d_wt   = (_Float16*)(base + OFF_WT);
  float* R0 = (float*)(base + OFF_R0);
  float* R1 = (float*)(base + OFF_R1);
  float* R2 = (float*)(base + OFF_R2);

  hipMemcpyAsync(d_filt, h_filt, (size_t)fpos * sizeof(float),
                 hipMemcpyHostToDevice, stream);

  // ---- synthesis input: prep -> features -> GEMM (weight/sqrt(512)) ----
  k_input_prep<<<gblk((long long)NB * C0), 256, 0, stream>>>(
      ws_in, iaw, iab, ifr, iph, d_prep, C0, sr0, bw0);
  k_feat<<<gblk((long long)NB * C0 * N0 * N0), 256, 0, stream>>>(
      d_prep, R1, C0, N0, sr0);
  {
    int CoPad = (C0 + 15) & ~15;
    int Kpad = (C0 + 31) & ~31;
    k_modw<<<dim3(CoPad, NB), 256, 0, stream>>>(iwt, nullptr, d_wt, C0, C0, 1,
                                                CoPad, Kpad, 0);
    int P = N0 * N0;
    k_conv_wmma<1><<<dim3((P + 63) / 64, 1, NB), 256, 0, stream>>>(
        R1, d_wt, nullptr, R0, C0, N0, N0, C0, CoPad, 0, N0, N0, Kpad);
  }

  // ---- layers: X lives in R0; conv->R1; up_h->R2; up_v->R1; down->R0 ----
  for (int i = 0; i < NLAYER; ++i) {
    const LCfg& c = L[i];
    k_styles<<<gblk((long long)NB * c.in_ch), 256, 0, stream>>>(
        ws_in, aff_w[i], aff_b[i], d_sty, c.in_ch, i + 1);
    int kk = c.k * c.k;
    int Ktot = c.in_ch * kk;
    int Kpad = (Ktot + 31) & ~31;
    int CoPad = (c.out_ch + 15) & ~15;
    k_modw<<<dim3(CoPad, NB), 256, 0, stream>>>(
        conv_w[i], d_sty, d_wt, c.out_ch, c.in_ch, kk, CoPad, Kpad,
        c.is_rgb ? 0 : 1);
    int Hc = c.in_size + (c.k == 3 ? 2 : 0);
    int P = Hc * Hc;
    if (c.k == 3) {
      k_conv_wmma<3><<<dim3((P + 63) / 64, 1, NB), 256, 0, stream>>>(
          R0, d_wt, conv_b[i], R1, c.in_ch, c.in_size, c.in_size, c.out_ch,
          CoPad, 2, Hc, Hc, Kpad);
    } else {
      k_conv_wmma<1><<<dim3((P + 63) / 64, 1, NB), 256, 0, stream>>>(
          R0, d_wt, conv_b[i], R1, c.in_ch, c.in_size, c.in_size, c.out_ch,
          CoPad, 0, Hc, Hc, Kpad);
    }
    if (c.is_rgb) {
      k_rgb_final<<<gblk((long long)NB * 3 * IMG * IMG), 256, 0, stream>>>(
          R1, (float*)d_out);
      continue;
    }
    int Hp = Hc * c.up + c.p0 + c.p1;
    int W1 = Hp - c.upT + 1;
    const float* fu = d_filt + c.fuOff;
    k_up_h<<<gblk((long long)NB * c.out_ch * Hp * W1), 256, 0, stream>>>(
        R1, R2, c.out_ch, Hc, Hc, c.up, c.p0, c.upT, fu, (float)c.up, Hp, W1);
    k_up_v_lrelu<<<gblk((long long)NB * c.out_ch * W1 * W1), 256, 0, stream>>>(
        R2, R1, c.out_ch, Hp, W1, c.upT, fu, (float)c.up, c.gain, c.slope);
    const float* fd = d_filt + c.fdOff;
    int Ho = c.out_size;
    if (!c.is_crit) {
      k_down_rad<<<gblk((long long)NB * c.out_ch * Ho * Ho), 256, 0, stream>>>(
          R1, R0, c.out_ch, W1, c.dnT, fd, c.down, Ho);
    } else {
      k_down_sep_h<<<gblk((long long)NB * c.out_ch * W1 * Ho), 256, 0, stream>>>(
          R1, R2, c.out_ch, W1, c.dnT, fd, c.down, Ho);
      k_down_sep_v<<<gblk((long long)NB * c.out_ch * Ho * Ho), 256, 0, stream>>>(
          R2, R0, c.out_ch, W1, Ho, c.dnT, fd, c.down, Ho);
    }
  }
}
